// SelfAttention_85804856639514
// MI455X (gfx1250) — matrix operand order repaired and checked by reading, not compiled, not run on hardware
//
#include <hip/hip_runtime.h>
#include <math.h>

#define BB 2
#define LL 2048
#define DD 1024
#define HH 16
#define DHD 64
#define MROWS (BB * LL) /* 4096 */

typedef __attribute__((ext_vector_type(16))) __bf16 v16bf;
typedef __attribute__((ext_vector_type(8)))  __bf16 v8bf;
typedef __attribute__((ext_vector_type(8)))  float  v8f;

// ---------- scalar helpers ----------
__device__ __forceinline__ unsigned short f2us(float f) {
  unsigned int u = __float_as_uint(f);
  u += 0x7FFFu + ((u >> 16) & 1u);         // round-to-nearest-even bf16
  return (unsigned short)(u >> 16);
}
__device__ __forceinline__ __bf16 us2bf(unsigned short s) {
  union { unsigned short u; __bf16 b; } x; x.u = s; return x.b;
}

// ---------- WMMA operand loaders (16-bit, wave32) ----------
// A-matrix 16x32: lane holds row (lane%16); K runs [g*8, g*8+8) and [16+g*8, ...)
__device__ __forceinline__ v16bf load_opA(const unsigned short* rowp, int g) {
  const unsigned short* p = rowp + g * 8;
  v8bf lo = *(const v8bf*)(const void*)(p);
  v8bf hi = *(const v8bf*)(const void*)(p + 16);
  v16bf r = __builtin_shufflevector(lo, hi, 0,1,2,3,4,5,6,7,8,9,10,11,12,13,14,15);
  return r;
}
// B-matrix 32x16: lane holds col (lane%16); K runs [g*8, g*8+8) and [16+g*8, ...)
__device__ __forceinline__ v16bf load_opB(const unsigned short* rowp, int g) {
  const unsigned short* p = rowp + g * 8;
  v8bf lo = *(const v8bf*)(const void*)(p);
  v8bf hi = *(const v8bf*)(const void*)(p + 16);
  v16bf r = __builtin_shufflevector(lo, hi, 0,1,2,3,4,5,6,7,8,9,10,11,12,13,14,15);
  return r;
}

__device__ __forceinline__ v8f wmma_bf16(v16bf a, v16bf b, v8f c) {
  return __builtin_amdgcn_wmma_f32_16x16x32_bf16(false, a, false, b, (short)0, c, false, false);
}

// ---------- fp32 -> bf16 conversion ----------
__global__ void __launch_bounds__(256) k_f32_to_bf16(const float* __restrict__ in,
                                                     unsigned short* __restrict__ out, int n) {
  int i = blockIdx.x * blockDim.x + threadIdx.x;
  int stride = gridDim.x * blockDim.x;
  for (; i < n; i += stride) out[i] = f2us(in[i]);
}

// ================= GEMM core: 32x64 tile per wave, ping-pong pipelined =================
// Computes acc[2][4] (two 16-row M-subtiles x four 16-col N-subtiles) of A * B^T,
// A rows from arow0/arow1, B rows from wrow[0..3], K = DD, no register copies.
struct GemmAcc { v8f a[2][4]; };

__device__ __forceinline__ GemmAcc gemm_32x64(const unsigned short* __restrict__ arow0,
                                              const unsigned short* __restrict__ arow1,
                                              const unsigned short* const* wrow, int g) {
  v8f vz = {0.f,0.f,0.f,0.f,0.f,0.f,0.f,0.f};
  GemmAcc r;
#pragma unroll
  for (int ms = 0; ms < 2; ++ms)
#pragma unroll
    for (int ns = 0; ns < 4; ++ns) r.a[ms][ns] = vz;

  // ping-pong operand buffers: no copies, roles alternate
  v16bf a0A, a1A, bA[4], a0B, a1B, bB[4];

  a0A = load_opA(arow0, g);
  a1A = load_opA(arow1, g);
#pragma unroll
  for (int ns = 0; ns < 4; ++ns) bA[ns] = load_opB(wrow[ns], g);

#pragma unroll
  for (int k = 0; k < DD - 64; k += 64) {
    // fill B-buffer for k+32 while computing on A-buffer
    a0B = load_opA(arow0 + k + 32, g);
    a1B = load_opA(arow1 + k + 32, g);
#pragma unroll
    for (int ns = 0; ns < 4; ++ns) bB[ns] = load_opB(wrow[ns] + k + 32, g);
#pragma unroll
    for (int ns = 0; ns < 4; ++ns) r.a[0][ns] = wmma_bf16(a0A, bA[ns], r.a[0][ns]);
#pragma unroll
    for (int ns = 0; ns < 4; ++ns) r.a[1][ns] = wmma_bf16(a1A, bA[ns], r.a[1][ns]);

    // fill A-buffer for k+64 while computing on B-buffer
    a0A = load_opA(arow0 + k + 64, g);
    a1A = load_opA(arow1 + k + 64, g);
#pragma unroll
    for (int ns = 0; ns < 4; ++ns) bA[ns] = load_opB(wrow[ns] + k + 64, g);
#pragma unroll
    for (int ns = 0; ns < 4; ++ns) r.a[0][ns] = wmma_bf16(a0B, bB[ns], r.a[0][ns]);
#pragma unroll
    for (int ns = 0; ns < 4; ++ns) r.a[1][ns] = wmma_bf16(a1B, bB[ns], r.a[1][ns]);
  }

  // tail pair: k = DD-64 (A-buffer holds it), k = DD-32 into B-buffer
  a0B = load_opA(arow0 + DD - 32, g);
  a1B = load_opA(arow1 + DD - 32, g);
#pragma unroll
  for (int ns = 0; ns < 4; ++ns) bB[ns] = load_opB(wrow[ns] + DD - 32, g);
#pragma unroll
  for (int ns = 0; ns < 4; ++ns) r.a[0][ns] = wmma_bf16(a0A, bA[ns], r.a[0][ns]);
#pragma unroll
  for (int ns = 0; ns < 4; ++ns) r.a[1][ns] = wmma_bf16(a1A, bA[ns], r.a[1][ns]);
#pragma unroll
  for (int ns = 0; ns < 4; ++ns) r.a[0][ns] = wmma_bf16(a0B, bB[ns], r.a[0][ns]);
#pragma unroll
  for (int ns = 0; ns < 4; ++ns) r.a[1][ns] = wmma_bf16(a1B, bB[ns], r.a[1][ns]);
  return r;
}

// ---------- QKV projection: qkv = x @ w_in^T + b_in, scatter to (B,H,L,DH) bf16 ----------
__global__ void __launch_bounds__(256)
k_qkv_gemm(const unsigned short* __restrict__ xbf,   // (MROWS, DD) bf16
           const unsigned short* __restrict__ wbf,   // (3*DD, DD) bf16
           const float* __restrict__ bias,           // (3*DD,)
           unsigned short* __restrict__ qb,          // (B*H, L, DH) bf16 (q pre-scaled)
           unsigned short* __restrict__ kb,
           unsigned short* __restrict__ vb) {
  const int lane = threadIdx.x & 31, wid = threadIdx.x >> 5;
  const int g = lane >> 4, c = lane & 15;
  const int wg = blockIdx.x * 8 + wid;               // 6144 waves
  const int mt = wg / 48, nt = wg % 48;              // 128 x 48 tiles
  const int m0 = mt * 32, n0 = nt * 64;

  const unsigned short* xrow0 = xbf + (size_t)(m0 + c) * DD;
  const unsigned short* xrow1 = xbf + (size_t)(m0 + 16 + c) * DD;
  const unsigned short* wrow[4];
#pragma unroll
  for (int ns = 0; ns < 4; ++ns) wrow[ns] = wbf + (size_t)(n0 + 16 * ns + c) * DD;

  GemmAcc acc = gemm_32x64(xrow0, xrow1, wrow, g);

#pragma unroll
  for (int ms = 0; ms < 2; ++ms) {
#pragma unroll
    for (int ns = 0; ns < 4; ++ns) {
      const int col = n0 + 16 * ns + c;
      const int sel = col >> 10, wi = col & 1023;
      const int head = wi >> 6, dh = wi & 63;
      unsigned short* dst = (sel == 0) ? qb : ((sel == 1) ? kb : vb);
      const float scl = (sel == 0) ? 0.125f : 1.0f;  // fold 1/sqrt(64) into q
      const float bv = bias[col];
#pragma unroll
      for (int v = 0; v < 8; ++v) {
        const int row = m0 + 16 * ms + v + 8 * g;    // global row in (B*L)
        const int bidx = row >> 11, l = row & 2047;
        const size_t di = ((size_t)(bidx * HH + head) * LL + l) * DHD + dh;
        dst[di] = f2us((acc.a[ms][ns][v] + bv) * scl);
      }
    }
  }
}

// ---------- flash attention: one wave per 16-query tile per (b,h) ----------
__global__ void __launch_bounds__(128)
k_flash_attn(const unsigned short* __restrict__ qb,
             const unsigned short* __restrict__ kbuf,
             const unsigned short* __restrict__ vbuf,
             const int* __restrict__ cmask,
             unsigned short* __restrict__ ao) {      // (B*L, DD) bf16
  __shared__ alignas(16) unsigned short Pl[4][16 * 32];
  __shared__ alignas(16) unsigned short Vl[4][32 * 72]; // padded stride 72

  const int lane = threadIdx.x & 31, wid = threadIdx.x >> 5;
  const int g = lane >> 4, c = lane & 15;
  const int wg = blockIdx.x * 4 + wid;               // 4096 waves
  const int bh = wg >> 7;                            // 128 q-tiles per head
  const int qt = wg & 127;
  const int m0 = qt * 16;
  const int causal = cmask[0];

  const unsigned short* q  = qb   + (size_t)bh * LL * DHD;
  const unsigned short* kp = kbuf + (size_t)bh * LL * DHD;
  const unsigned short* vp = vbuf + (size_t)bh * LL * DHD;
  unsigned short* Pw = Pl[wid];
  unsigned short* Vw = Vl[wid];

  // per-lane LDS byte address for async DMA of this lane's V row
  const unsigned int vlds = (unsigned int)(unsigned long long)(void*)(&Vw[lane * 72]);

  const v16bf aQ0 = load_opA(q + (size_t)(m0 + c) * DHD + 0,  g);
  const v16bf aQ1 = load_opA(q + (size_t)(m0 + c) * DHD + 32, g);

  v8f vz = {0.f,0.f,0.f,0.f,0.f,0.f,0.f,0.f};
  v8f o[4] = {vz, vz, vz, vz};
  float m_run[8], l_run[8];
#pragma unroll
  for (int v = 0; v < 8; ++v) { m_run[v] = -1e30f; l_run[v] = 0.f; }

  const int kbmax = causal ? ((m0 + 15) >> 5) : (LL / 32 - 1);

  for (int kb = 0; kb <= kbmax; ++kb) {
    const int n0 = kb * 32;

    // ---- kick off async DMA of the V block (32 x 64 bf16, lane -> its row) ----
    // overlaps with the QK^T WMMAs below; tracked by ASYNCcnt.
    {
      const unsigned long long vga =
          (unsigned long long)(const void*)(vp + (size_t)(n0 + lane) * DHD);
      asm volatile(
        "global_load_async_to_lds_b128 %0, %1, off\n\t"
        "global_load_async_to_lds_b128 %0, %1, off offset:16\n\t"
        "global_load_async_to_lds_b128 %0, %1, off offset:32\n\t"
        "global_load_async_to_lds_b128 %0, %1, off offset:48\n\t"
        "global_load_async_to_lds_b128 %0, %1, off offset:64\n\t"
        "global_load_async_to_lds_b128 %0, %1, off offset:80\n\t"
        "global_load_async_to_lds_b128 %0, %1, off offset:96\n\t"
        "global_load_async_to_lds_b128 %0, %1, off offset:112"
        :: "v"(vlds), "v"(vga) : "memory");
    }

    // S = Q K^T  (two 16x16 N-tiles, K-dim = DH = 64 -> 2 WMMA steps each)
    v16bf bK00 = load_opB(kp + (size_t)(n0 + c) * DHD + 0,       g);
    v16bf bK01 = load_opB(kp + (size_t)(n0 + c) * DHD + 32,      g);
    v16bf bK10 = load_opB(kp + (size_t)(n0 + 16 + c) * DHD + 0,  g);
    v16bf bK11 = load_opB(kp + (size_t)(n0 + 16 + c) * DHD + 32, g);
    v8f S0 = wmma_bf16(aQ1, bK01, wmma_bf16(aQ0, bK00, vz));
    v8f S1 = wmma_bf16(aQ1, bK11, wmma_bf16(aQ0, bK10, vz));

    // online softmax (C layout: vgpr v <-> row v+8g, lane%16 <-> col)
    float mnew[8], alpha[8];
#pragma unroll
    for (int v = 0; v < 8; ++v) {
      float s0 = S0[v], s1 = S1[v];
      if (causal) {
        const int row = m0 + v + 8 * g;
        if (n0 + c      > row) s0 = -1e30f;
        if (n0 + 16 + c > row) s1 = -1e30f;
      }
      S0[v] = s0; S1[v] = s1;
      float t = fmaxf(s0, s1);
      t = fmaxf(t, __shfl_xor(t, 1, 32));
      t = fmaxf(t, __shfl_xor(t, 2, 32));
      t = fmaxf(t, __shfl_xor(t, 4, 32));
      t = fmaxf(t, __shfl_xor(t, 8, 32));
      const float mn = fmaxf(m_run[v], t);
      alpha[v] = __expf(m_run[v] - mn);
      m_run[v] = mn; mnew[v] = mn;
    }
#pragma unroll
    for (int v = 0; v < 8; ++v) {
      const float p0 = __expf(S0[v] - mnew[v]);
      const float p1 = __expf(S1[v] - mnew[v]);
      const int row = v + 8 * g;
      Pw[row * 32 + c]      = f2us(p0);
      Pw[row * 32 + 16 + c] = f2us(p1);
      float rs = p0 + p1;
      rs += __shfl_xor(rs, 1, 32);
      rs += __shfl_xor(rs, 2, 32);
      rs += __shfl_xor(rs, 4, 32);
      rs += __shfl_xor(rs, 8, 32);
      l_run[v] = l_run[v] * alpha[v] + rs;
#pragma unroll
      for (int cc = 0; cc < 4; ++cc) o[cc][v] *= alpha[v];
    }

    // wait for the V DMA to land in LDS before the PV gathers
    asm volatile("s_wait_asynccnt 0x0" ::: "memory");

    // O += P V : P as A-operand from LDS, V columns as B-operand from LDS
    const v16bf aP = load_opA(Pw + c * 32, g);
#pragma unroll
    for (int cc = 0; cc < 4; ++cc) {
      const int d = cc * 16 + c;
      v16bf bV;
#pragma unroll
      for (int i = 0; i < 16; ++i)
        bV[i] = us2bf(Vw[(g * 8 + (i & 7) + (i >> 3) * 16) * 72 + d]);
      o[cc] = wmma_bf16(aP, bV, o[cc]);
    }
  }

  // normalize + store bf16 attention output in (B*L, D) layout
  const int b = bh >> 4, head = bh & 15;
  float inv[8];
#pragma unroll
  for (int v = 0; v < 8; ++v) inv[v] = 1.0f / l_run[v];
#pragma unroll
  for (int cc = 0; cc < 4; ++cc) {
#pragma unroll
    for (int v = 0; v < 8; ++v) {
      const int row_l = m0 + v + 8 * g;
      const size_t di = ((size_t)(b * LL + row_l)) * DD + head * DHD + cc * 16 + c;
      ao[di] = f2us(o[cc][v] * inv[v]);
    }
  }
}

// ---------- output projection: out = ao @ w_out^T (fp32 out), 32x64 tile ----------
__global__ void __launch_bounds__(256)
k_out_gemm(const unsigned short* __restrict__ abf,   // (MROWS, DD) bf16
           const unsigned short* __restrict__ wbf,   // (DD, DD) bf16
           float* __restrict__ out) {
  const int lane = threadIdx.x & 31, wid = threadIdx.x >> 5;
  const int g = lane >> 4, c = lane & 15;
  const int wg = blockIdx.x * 8 + wid;               // 2048 waves
  const int mt = wg >> 4, nt = wg & 15;              // 128 x 16 tiles
  const int m0 = mt * 32, n0 = nt * 64;

  const unsigned short* arow0 = abf + (size_t)(m0 + c) * DD;
  const unsigned short* arow1 = abf + (size_t)(m0 + 16 + c) * DD;
  const unsigned short* wrow[4];
#pragma unroll
  for (int ns = 0; ns < 4; ++ns) wrow[ns] = wbf + (size_t)(n0 + 16 * ns + c) * DD;

  GemmAcc acc = gemm_32x64(arow0, arow1, wrow, g);

#pragma unroll
  for (int ms = 0; ms < 2; ++ms) {
#pragma unroll
    for (int ns = 0; ns < 4; ++ns) {
      const int col = n0 + 16 * ns + c;
#pragma unroll
      for (int v = 0; v < 8; ++v) {
        const int row = m0 + 16 * ms + v + 8 * g;
        out[(size_t)row * DD + col] = acc.a[ms][ns][v];
      }
    }
  }
}

// ---------- launcher ----------
extern "C" void kernel_launch(void* const* d_in, const int* in_sizes, int n_in,
                              void* d_out, int out_size, void* d_ws, size_t ws_size,
                              hipStream_t stream) {
  (void)in_sizes; (void)n_in; (void)out_size;
  const float* x     = (const float*)d_in[0];
  const float* w_in  = (const float*)d_in[1];
  const float* b_in  = (const float*)d_in[2];
  const float* w_out = (const float*)d_in[3];
  const int*   cmask = (const int*)d_in[4];

  // workspace layout (bytes), all 16B-aligned
  char* ws = (char*)d_ws;
  const size_t oXBF  = 0;                                   // 8 MB
  const size_t oWIN  = oXBF  + (size_t)MROWS * DD * 2;      // 6 MB
  const size_t oWOUT = oWIN  + (size_t)3 * DD * DD * 2;     // 2 MB
  const size_t oQ    = oWOUT + (size_t)DD * DD * 2;         // 8 MB
  const size_t oK    = oQ    + (size_t)MROWS * DD * 2;
  const size_t oV    = oK    + (size_t)MROWS * DD * 2;
  const size_t oAO   = oV    + (size_t)MROWS * DD * 2;
  const size_t total = oAO   + (size_t)MROWS * DD * 2;      // 48 MB
  if (ws_size < total) return;

  unsigned short* xbf    = (unsigned short*)(ws + oXBF);
  unsigned short* winbf  = (unsigned short*)(ws + oWIN);
  unsigned short* woutbf = (unsigned short*)(ws + oWOUT);
  unsigned short* qb     = (unsigned short*)(ws + oQ);
  unsigned short* kb     = (unsigned short*)(ws + oK);
  unsigned short* vb     = (unsigned short*)(ws + oV);
  unsigned short* ao     = (unsigned short*)(ws + oAO);

  k_f32_to_bf16<<<1024, 256, 0, stream>>>(x,     xbf,    MROWS * DD);
  k_f32_to_bf16<<<1024, 256, 0, stream>>>(w_in,  winbf,  3 * DD * DD);
  k_f32_to_bf16<<<512,  256, 0, stream>>>(w_out, woutbf, DD * DD);

  k_qkv_gemm<<<768, 256, 0, stream>>>(xbf, winbf, b_in, qb, kb, vb);
  k_flash_attn<<<1024, 128, 0, stream>>>(qb, kb, vb, cmask, ao);
  k_out_gemm<<<256, 256, 0, stream>>>(ao, woutbf, (float*)d_out);
}
